// MTAttention_v23_28321014349905
// MI455X (gfx1250) — matrix operand order repaired and checked by reading, not compiled, not run on hardware
//
#include <hip/hip_runtime.h>
#include <hip/hip_bf16.h>
#include <math.h>

// ---------------- problem constants ----------------
#define BATCH   64
#define CIN     512
#define CCH     512
#define HDIM    31
#define NPIX    961            // 31*31
#define HEADS   8
#define DHEAD   64
#define MROWS   61504          // BATCH*NPIX
#define NCOLS   1024           // [v | k] concatenated columns
#define NWIN    75
#define NROWS2  76             // 1 + 75
#define MFC     4864           // BATCH*NROWS2
#define KFC     1024           // CIN + CCH
#define CENTER  480            // 15*31+15

static constexpr float INV_MROWS = 1.0f / 61504.0f;
static constexpr float INV_MFC   = 1.0f / 4864.0f;

typedef __attribute__((ext_vector_type(16))) __bf16 v16bf;
typedef __attribute__((ext_vector_type(8)))  float  v8f;
typedef __attribute__((ext_vector_type(2)))  __bf16 bf16x2;
typedef __attribute__((ext_vector_type(4)))  unsigned int u32x4;
typedef __attribute__((ext_vector_type(8)))  int  i32x8;
typedef __attribute__((ext_vector_type(4)))  int  i32x4;

__device__ __forceinline__ float bflo(unsigned u) { return __uint_as_float(u << 16); }
__device__ __forceinline__ float bfhi(unsigned u) { return __uint_as_float(u & 0xffff0000u); }

// =====================================================================
// Kernel 1: vk_pre = flat @ [w_v | w_k] + [b_v | b_k]   (bf16 out, WMMA)
// 256 threads = 8 waves; WG tile 64(M) x 128(N); K step 32.
// Double-buffered LDS: one barrier per K-step, loads overlap WMMA.
// =====================================================================
__global__ __launch_bounds__(256) void gemm_vk_kernel(
    const float* __restrict__ x,
    const float* __restrict__ wv, const float* __restrict__ wk,
    const float* __restrict__ bv, const float* __restrict__ bk,
    __bf16* __restrict__ vkpre)
{
    __shared__ __bf16 As[2][64][34];    // [buf][m_sub][k_sub]
    __shared__ __bf16 Bs[2][128][34];   // [buf][n_sub][k_sub]

    const int m0  = blockIdx.x * 64;     // 961 blocks
    const int n0  = blockIdx.y * 128;    // 8 blocks
    const int tid = threadIdx.x;
    const int wave = tid >> 5, lane = tid & 31;
    const int l = lane & 15, hi = lane >> 4;
    const int mt = wave & 3;             // wave M tile
    const int ng = wave >> 2;            // wave N group (0/1) of 4 tiles

    // A-tile geometry: 1024 bf16-pairs, 4 per thread (coalesced over m)
    int amsub[4], akp[4]; size_t abase[4];
#pragma unroll
    for (int i = 0; i < 4; ++i) {
        int idx  = tid + i * 256;
        int msub = idx & 63;
        int kp   = idx >> 6;             // pair index 0..15
        int m  = m0 + msub;
        int bb = m / NPIX;
        int nn = m - bb * NPIX;
        amsub[i] = msub; akp[i] = kp;
        abase[i] = ((size_t)bb * CIN) * NPIX + nn;
    }
    // B-tile geometry: each thread owns one column, 16 adjacent k values
    const int bnc  = tid & 127;
    const int bhal = tid >> 7;
    const int bcol = n0 + bnc;
    const float* wp = (bcol < CCH) ? (wv + bcol) : (wk + (bcol - CCH));

    auto loadA = [&](int k0, int buf) {
#pragma unroll
        for (int i = 0; i < 4; ++i) {
            int k = k0 + akp[i] * 2;
            float f0 = x[abase[i] + (size_t)k * NPIX];
            float f1 = x[abase[i] + (size_t)(k + 1) * NPIX];
            bf16x2 p; p[0] = (__bf16)f0; p[1] = (__bf16)f1;
            *(bf16x2*)&As[buf][amsub[i]][akp[i] * 2] = p;
        }
    };
    auto loadB = [&](int k0, int buf) {
#pragma unroll
        for (int j = 0; j < 16; j += 2) {
            int ks = bhal * 16 + j;
            float f0 = wp[(size_t)(k0 + ks) * CCH];
            float f1 = wp[(size_t)(k0 + ks + 1) * CCH];
            bf16x2 p; p[0] = (__bf16)f0; p[1] = (__bf16)f1;
            *(bf16x2*)&Bs[buf][bnc][ks] = p;
        }
    };

    v8f acc[4];
#pragma unroll
    for (int a = 0; a < 4; ++a) acc[a] = (v8f){0.f,0.f,0.f,0.f,0.f,0.f,0.f,0.f};

    loadA(0, 0); loadB(0, 0);
    __syncthreads();

    const int KT = CIN / 32;             // 16 steps
    for (int kt = 0; kt < KT; ++kt) {
        const int cur = kt & 1, nxt = cur ^ 1;
        if (kt + 1 < KT) { loadA((kt + 1) * 32, nxt); loadB((kt + 1) * 32, nxt); }

        v16bf afrag;
        const int mrow = mt * 16 + l;
#pragma unroll
        for (int j = 0; j < 16; ++j) {
            int v  = j >> 1;
            int kk = (v < 4) ? (2 * v + (j & 1) + hi * 8)
                             : (16 + 2 * (v - 4) + (j & 1) + hi * 8);
            afrag[j] = As[cur][mrow][kk];
        }
#pragma unroll
        for (int a = 0; a < 4; ++a) {
            v16bf bfrag;
            const int nrow = ng * 64 + a * 16 + l;
#pragma unroll
            for (int j = 0; j < 16; ++j)
                bfrag[j] = Bs[cur][nrow][(j < 8) ? (hi * 8 + j) : (16 + hi * 8 + (j - 8))];
            acc[a] = __builtin_amdgcn_wmma_f32_16x16x32_bf16(
                         false, afrag, false, bfrag, (short)0, acc[a], false, false);
        }
        __syncthreads();
    }

    const float* bias_base = (n0 < CCH) ? bv : (bk - CCH);  // block-uniform
#pragma unroll
    for (int a = 0; a < 4; ++a) {
        const int col = n0 + ng * 64 + a * 16 + l;
        const float bias = bias_base[col];
#pragma unroll
        for (int v = 0; v < 8; ++v) {
            int row = m0 + mt * 16 + hi * 8 + v;
            vkpre[(size_t)row * NCOLS + col] = (__bf16)(acc[a][v] + bias);
        }
    }
}

// =====================================================================
// Kernel 2: per-column sums / sumsq over bf16 matrix (paired columns)
// =====================================================================
__global__ __launch_bounds__(256) void colstats_bf16_kernel(
    const __bf16* __restrict__ a, int rows, int colpairs,
    float* __restrict__ sums, float* __restrict__ sumsq)
{
    int cp = blockIdx.x * blockDim.x + threadIdx.x;
    if (cp >= colpairs) return;
    const unsigned* pa = (const unsigned*)a;
    float s0 = 0.f, q0 = 0.f, s1 = 0.f, q1 = 0.f;
    for (int r = 0; r < rows; ++r) {
        unsigned u = pa[(size_t)r * colpairs + cp];
        float lo = bflo(u), hi = bfhi(u);
        s0 += lo; q0 += lo * lo;
        s1 += hi; q1 += hi * hi;
    }
    sums[2 * cp]     = s0; sumsq[2 * cp]     = q0;
    sums[2 * cp + 1] = s1; sumsq[2 * cp + 1] = q1;
}

__global__ __launch_bounds__(256) void colstats_f32_kernel(
    const float* __restrict__ a, int rows, int cols,
    float* __restrict__ sums, float* __restrict__ sumsq)
{
    int c = blockIdx.x * blockDim.x + threadIdx.x;
    if (c >= cols) return;
    float s = 0.f, s2 = 0.f;
    for (int r = 0; r < rows; ++r) {
        float v = a[(size_t)r * cols + c];
        s += v; s2 += v * v;
    }
    sums[c] = s; sumsq[c] = s2;
}

// =====================================================================
// Kernel 3: per-(b,h) attention: BN(k) on the fly, q.k, softmax over 961
// =====================================================================
__global__ __launch_bounds__(256) void attn_softmax_kernel(
    const __bf16* __restrict__ vkpre,
    const float* __restrict__ sums, const float* __restrict__ sumsq,
    const float* __restrict__ g_kn, const float* __restrict__ b_kn,
    float* __restrict__ attn)
{
    const int b = blockIdx.x >> 3;
    const int h = blockIdx.x & 7;
    const int tid = threadIdx.x;

    __shared__ float qs[DHEAD], ksc[DHEAD], ksh[DHEAD];
    __shared__ float sc[NPIX];
    __shared__ float red[256];

    if (tid < DHEAD) {
        int cc  = h * DHEAD + tid;
        int col = CCH + cc;
        float mu  = sums[col] * INV_MROWS;
        float var = sumsq[col] * INV_MROWS - mu * mu;
        float s   = g_kn[cc] * rsqrtf(var + 1e-5f);
        ksc[tid] = s;
        ksh[tid] = b_kn[cc] - mu * s;
    }
    __syncthreads();
    if (tid < DHEAD) {
        size_t base = ((size_t)(b * NPIX + CENTER)) * NCOLS + CCH + h * DHEAD;
        qs[tid] = (float)vkpre[base + tid] * ksc[tid] + ksh[tid];
    }
    __syncthreads();

    float lmax = -1e30f;
    for (int n = tid; n < NPIX; n += 256) {
        size_t base = ((size_t)(b * NPIX + n)) * NCOLS + CCH + h * DHEAD;
        const uint4* p4 = (const uint4*)(vkpre + base);  // 128B aligned
        float dot = 0.f;
#pragma unroll
        for (int t = 0; t < 8; ++t) {
            uint4 u = p4[t];
            int dd = t * 8;
            dot += qs[dd + 0] * (bflo(u.x) * ksc[dd + 0] + ksh[dd + 0]);
            dot += qs[dd + 1] * (bfhi(u.x) * ksc[dd + 1] + ksh[dd + 1]);
            dot += qs[dd + 2] * (bflo(u.y) * ksc[dd + 2] + ksh[dd + 2]);
            dot += qs[dd + 3] * (bfhi(u.y) * ksc[dd + 3] + ksh[dd + 3]);
            dot += qs[dd + 4] * (bflo(u.z) * ksc[dd + 4] + ksh[dd + 4]);
            dot += qs[dd + 5] * (bfhi(u.z) * ksc[dd + 5] + ksh[dd + 5]);
            dot += qs[dd + 6] * (bflo(u.w) * ksc[dd + 6] + ksh[dd + 6]);
            dot += qs[dd + 7] * (bfhi(u.w) * ksc[dd + 7] + ksh[dd + 7]);
        }
        float s = dot * 0.125f;
        sc[n] = s;
        lmax = fmaxf(lmax, s);
    }
    red[tid] = lmax; __syncthreads();
    for (int st = 128; st > 0; st >>= 1) {
        if (tid < st) red[tid] = fmaxf(red[tid], red[tid + st]);
        __syncthreads();
    }
    const float mx = red[0];
    __syncthreads();

    float lsum = 0.f;
    for (int n = tid; n < NPIX; n += 256) {
        float e = __expf(sc[n] - mx);
        sc[n] = e; lsum += e;
    }
    red[tid] = lsum; __syncthreads();
    for (int st = 128; st > 0; st >>= 1) {
        if (tid < st) red[tid] += red[tid + st];
        __syncthreads();
    }
    const float inv = 1.0f / red[0];
    __syncthreads();

    float* ap = attn + (size_t)(b * HEADS + h) * NPIX;
    for (int n = tid; n < NPIX; n += 256) ap[n] = sc[n] * inv;
}

// ---- shared helper: window coordinates (w in [0,75)) ----
__device__ __forceinline__ void win_coords(int w, int& r0, int& r1,
                                           int& c0, int& c1)
{
    const int m = HDIM / 2;
    int i = w / 5 + 1, j = w % 5;
    switch (j) {
        case 0:  r0 = m - i; r1 = m + i + 1; c0 = m - i; c1 = m + i + 1; break;
        case 1:  r0 = m - i; r1 = m + 1;     c0 = m - i; c1 = m + i + 1; break;
        case 2:  r0 = m;     r1 = m + i + 1; c0 = m - i; c1 = m + i + 1; break;
        case 3:  r0 = m - i; r1 = m + i + 1; c0 = m - i; c1 = m + 1;     break;
        default: r0 = m - i; r1 = m + i + 1; c0 = m;     c1 = m + i + 1; break;
    }
}

// =====================================================================
// Kernel 4: per-(b,h) integral of attn, 75 window means -> ma
// =====================================================================
__global__ __launch_bounds__(256) void ma_calc_kernel(
    const float* __restrict__ attn, float* __restrict__ ma)
{
    __shared__ float I[32][33];
    const int b = blockIdx.x >> 3, h = blockIdx.x & 7;
    const int tid = threadIdx.x;
    const float* a = attn + (size_t)(b * HEADS + h) * NPIX;

    for (int idx = tid; idx < 32 * 32; idx += 256) {
        int r = idx >> 5, c = idx & 31;
        I[r][c] = (r > 0 && c > 0) ? a[(r - 1) * HDIM + (c - 1)] : 0.0f;
    }
    __syncthreads();
    if (tid >= 1 && tid < 32) {
        float s = 0.f;
        for (int c = 1; c < 32; ++c) { s += I[tid][c]; I[tid][c] = s; }
    }
    __syncthreads();
    if (tid >= 1 && tid < 32) {
        float s = 0.f;
        for (int r = 1; r < 32; ++r) { s += I[r][tid]; I[r][tid] = s; }
    }
    __syncthreads();
    if (tid < NWIN) {
        int r0, r1, c0, c1; win_coords(tid, r0, r1, c0, c1);
        float cnt = (float)((r1 - r0) * (c1 - c0));
        float box = I[r1][c1] - I[r0][c1] - I[r1][c0] + I[r0][c0];
        ma[(size_t)(b * HEADS + h) * NWIN + tid] = box / cnt;
    }
}

// =====================================================================
// Kernel 5: xw = attn * relu(BN(v)); integral + windows -> y
// Block = (8 channels, b). Each thread reads one uint4 (8 channels) per
// pixel and scatters to 8 LDS integral images.
// =====================================================================
__global__ __launch_bounds__(256) void xw_y_kernel(
    const __bf16* __restrict__ vkpre,
    const float* __restrict__ sums, const float* __restrict__ sumsq,
    const float* __restrict__ g_vn, const float* __restrict__ b_vn,
    const float* __restrict__ attn, const float* __restrict__ ma,
    float* __restrict__ y)
{
    __shared__ float I[8][32][33];
    __shared__ float sc8[8], sh8[8], xwc8[8];
    const int cblk = blockIdx.x;        // 64 blocks of 8 channels
    const int b    = blockIdx.y;
    const int tid  = threadIdx.x;
    const int cbase = cblk * 8;
    const int h     = cbase >> 6;       // all 8 channels share one head

    if (tid < 8) {
        int c = cbase + tid;
        float mu  = sums[c] * INV_MROWS;
        float var = sumsq[c] * INV_MROWS - mu * mu;
        float s   = g_vn[c] * rsqrtf(var + 1e-5f);
        sc8[tid] = s; sh8[tid] = b_vn[c] - mu * s;
    }
    for (int idx = tid; idx < 8 * 32; idx += 256) {
        int ch = idx >> 5, rc = idx & 31;
        I[ch][0][rc] = 0.f;
        I[ch][rc][0] = 0.f;
    }
    __syncthreads();

    const float* at = attn + (size_t)(b * HEADS + h) * NPIX;
    for (int n = tid; n < NPIX; n += 256) {
        uint4 u = *(const uint4*)(vkpre + ((size_t)(b * NPIX + n)) * NCOLS + cbase);
        float vals[8];
        vals[0] = bflo(u.x); vals[1] = bfhi(u.x);
        vals[2] = bflo(u.y); vals[3] = bfhi(u.y);
        vals[4] = bflo(u.z); vals[5] = bfhi(u.z);
        vals[6] = bflo(u.w); vals[7] = bfhi(u.w);
        int r = n / HDIM, cc = n - r * HDIM;
        float a_n = at[n];
#pragma unroll
        for (int ch = 0; ch < 8; ++ch) {
            float vv = fmaxf(vals[ch] * sc8[ch] + sh8[ch], 0.f) * a_n;
            I[ch][r + 1][cc + 1] = vv;
            if (n == CENTER) xwc8[ch] = vv;
        }
    }
    __syncthreads();
    if (tid < 8 * 31) {                 // row prefix per channel
        int ch = tid / 31, r = tid % 31 + 1;
        float ss = 0.f;
        for (int cc = 1; cc < 32; ++cc) { ss += I[ch][r][cc]; I[ch][r][cc] = ss; }
    }
    __syncthreads();
    if (tid < 8 * 31) {                 // column prefix per channel
        int ch = tid / 31, cc = tid % 31 + 1;
        float ss = 0.f;
        for (int r = 1; r < 32; ++r) { ss += I[ch][r][cc]; I[ch][r][cc] = ss; }
    }
    __syncthreads();
    if (tid < 8)
        y[((size_t)b * NROWS2) * CCH + cbase + tid] = xwc8[tid];
    for (int idx = tid; idx < 8 * NWIN; idx += 256) {
        int ch = idx / NWIN, w = idx - ch * NWIN;
        int r0, r1, c0, c1; win_coords(w, r0, r1, c0, c1);
        float cnt = (float)((r1 - r0) * (c1 - c0));
        float box = I[ch][r1][c1] - I[ch][r0][c1] - I[ch][r1][c0] + I[ch][r0][c0];
        float mav = ma[(size_t)(b * HEADS + h) * NWIN + w];
        y[((size_t)(b * NROWS2) + 1 + w) * CCH + cbase + ch] = (box / cnt) / (mav + 1e-16f);
    }
}

// =====================================================================
// Kernel 6a: one-shot strided gather of x0 = x[:, :, center] -> bf16
// =====================================================================
__global__ __launch_bounds__(256) void x0_extract_kernel(
    const float* __restrict__ x, __bf16* __restrict__ x0bf)
{
    int idx = blockIdx.x * 256 + threadIdx.x;    // b*CIN + c
    if (idx >= BATCH * CIN) return;
    x0bf[idx] = (__bf16)x[(size_t)idx * NPIX + CENTER];
}

// =====================================================================
// Kernel 6b: row L2-normalize y; assemble yy = [x0 | y/||y||] in bf16
// =====================================================================
__global__ __launch_bounds__(256) void ynorm_yy_kernel(
    const __bf16* __restrict__ x0bf, const float* __restrict__ y,
    __bf16* __restrict__ yy)
{
    const int b  = blockIdx.x / NROWS2;
    const int nr = blockIdx.x % NROWS2;
    const int tid = threadIdx.x;
    __shared__ float red[256];

    const float* yr = y + ((size_t)(b * NROWS2) + nr) * CCH;
    float ss = 0.f;
    for (int c = tid; c < CCH; c += 256) { float v = yr[c]; ss += v * v; }
    red[tid] = ss; __syncthreads();
    for (int st = 128; st > 0; st >>= 1) {
        if (tid < st) red[tid] += red[tid + st];
        __syncthreads();
    }
    const float inv = 1.0f / fmaxf(sqrtf(red[0]), 1e-12f);

    __bf16* out = yy + ((size_t)(b * NROWS2) + nr) * KFC;
    const unsigned* xs = (const unsigned*)(x0bf + (size_t)b * CIN);
    unsigned* outu = (unsigned*)out;
    for (int cp = tid; cp < CIN / 2; cp += 256) outu[cp] = xs[cp];
    for (int cp = tid; cp < CCH / 2; cp += 256) {
        bf16x2 p;
        p[0] = (__bf16)(yr[2 * cp] * inv);
        p[1] = (__bf16)(yr[2 * cp + 1] * inv);
        *(bf16x2*)&out[CIN + 2 * cp] = p;
    }
}

// =====================================================================
// Kernel 7: fc_pre = yy(4864x1024 bf16) @ w_fc + b_fc   (WMMA + TDM)
// A-tiles fetched by the Tensor Data Mover, double-buffered so the DMA
// of tile k+1 overlaps the WMMAs of tile k.
// =====================================================================
__global__ __launch_bounds__(256) void gemm_fc_kernel(
    const __bf16* __restrict__ yy, const float* __restrict__ wfc,
    const float* __restrict__ bfc, float* __restrict__ fc)
{
    __shared__ __bf16 As[2][64 * 32];   // TDM destinations, [m][k] unpadded
    __shared__ __bf16 Bs[2][128][34];

    const int m0  = blockIdx.x * 64;    // 76 blocks
    const int n0  = blockIdx.y * 128;   // 4 blocks
    const int tid = threadIdx.x;
    const int wave = tid >> 5, lane = tid & 31;
    const int l = lane & 15, hi = lane >> 4;
    const int mt = wave & 3;
    const int ng = wave >> 2;

    const unsigned lds0 = (unsigned)(size_t)(void*)&As[0][0];
    const int bnc  = tid & 127;
    const int bhal = tid >> 7;
    const float* wp = wfc + (n0 + bnc);

    auto issueA = [&](int k0, int buf) {
        if (tid < 32) {
            // ---- Tensor DMA descriptor (D#) per CDNA5 ISA ----
            unsigned long long ga =
                (unsigned long long)(size_t)(yy + (size_t)m0 * KFC + k0);
            u32x4 g0 = { 1u,                               // count=1, user mode
                         lds0 + (unsigned)buf * 4096u,     // lds_addr
                         (unsigned)(ga & 0xffffffffu),
                         (unsigned)((ga >> 32) & 0x1ffffffu) | (2u << 30) };
            i32x8 g1 = { 0x10000,        // data_size=2B, no multicast
                         0x04000000,     // tensor_dim0 = 1024 (bits 79:48)
                         0x13000000,     // tensor_dim1 = 4864 (bits 111:80)
                         (32 << 16),     // tile_dim0 = 32  (bits 127:112)
                         64,             // tile_dim1 = 64  (bits 143:128)
                         1024,           // tensor_dim0_stride (bits 207:160)
                         0, 0 };
            i32x4 gz = { 0, 0, 0, 0 };
#if defined(__clang_major__) && (__clang_major__ >= 23)
            i32x8 g4 = { 0, 0, 0, 0, 0, 0, 0, 0 };
            __builtin_amdgcn_tensor_load_to_lds(g0, g1, gz, gz, g4, 0);
#else
            __builtin_amdgcn_tensor_load_to_lds(g0, g1, gz, gz, 0);
#endif
        }
    };
    auto loadB = [&](int k0, int buf) {
#pragma unroll
        for (int j = 0; j < 16; j += 2) {
            int ks = bhal * 16 + j;
            float f0 = wp[(size_t)(k0 + ks) * CCH];
            float f1 = wp[(size_t)(k0 + ks + 1) * CCH];
            bf16x2 p; p[0] = (__bf16)f0; p[1] = (__bf16)f1;
            *(bf16x2*)&Bs[buf][bnc][ks] = p;
        }
    };

    v8f acc[4];
#pragma unroll
    for (int a = 0; a < 4; ++a) acc[a] = (v8f){0.f,0.f,0.f,0.f,0.f,0.f,0.f,0.f};

    issueA(0, 0); loadB(0, 0);
    if (tid < 32) __builtin_amdgcn_s_wait_tensorcnt(0);
    __syncthreads();

    const int KT = KFC / 32;             // 32 steps
    for (int kt = 0; kt < KT; ++kt) {
        const int cur = kt & 1, nxt = cur ^ 1;
        if (kt + 1 < KT) { issueA((kt + 1) * 32, nxt); loadB((kt + 1) * 32, nxt); }

        v16bf afrag;
        const int mrow = mt * 16 + l;
#pragma unroll
        for (int j = 0; j < 16; ++j) {
            int v  = j >> 1;
            int kk = (v < 4) ? (2 * v + (j & 1) + hi * 8)
                             : (16 + 2 * (v - 4) + (j & 1) + hi * 8);
            afrag[j] = As[cur][mrow * 32 + kk];
        }
#pragma unroll
        for (int a = 0; a < 4; ++a) {
            v16bf bfrag;
            const int nrow = ng * 64 + a * 16 + l;
#pragma unroll
            for (int j = 0; j < 16; ++j)
                bfrag[j] = Bs[cur][nrow][(j < 8) ? (hi * 8 + j) : (16 + hi * 8 + (j - 8))];
            acc[a] = __builtin_amdgcn_wmma_f32_16x16x32_bf16(
                         false, afrag, false, bfrag, (short)0, acc[a], false, false);
        }
        if (tid < 32) __builtin_amdgcn_s_wait_tensorcnt(0);
        __syncthreads();
    }

#pragma unroll
    for (int a = 0; a < 4; ++a) {
        const int col = n0 + ng * 64 + a * 16 + l;
        const float bias = bfc[col];
#pragma unroll
        for (int v = 0; v < 8; ++v) {
            int row = m0 + mt * 16 + hi * 8 + v;
            fc[(size_t)row * CCH + col] = acc[a][v] + bias;
        }
    }
}

// =====================================================================
// Kernel 9: out = relu(BN(fc_pre))
// =====================================================================
__global__ __launch_bounds__(256) void bn_relu_out_kernel(
    const float* __restrict__ fc,
    const float* __restrict__ sums, const float* __restrict__ sumsq,
    const float* __restrict__ g, const float* __restrict__ bta,
    float* __restrict__ out)
{
    int idx = blockIdx.x * 256 + threadIdx.x;
    if (idx >= MFC * CCH) return;
    int c = idx & (CCH - 1);
    float mu  = sums[c] * INV_MFC;
    float var = sumsq[c] * INV_MFC - mu * mu;
    float s   = g[c] * rsqrtf(var + 1e-5f);
    out[idx]  = fmaxf((fc[idx] - mu) * s + bta[c], 0.f);
}

// =====================================================================
// Host launch
// =====================================================================
extern "C" void kernel_launch(void* const* d_in, const int* in_sizes, int n_in,
                              void* d_out, int out_size, void* d_ws, size_t ws_size,
                              hipStream_t stream)
{
    (void)in_sizes; (void)n_in; (void)out_size; (void)ws_size;
    const float* x     = (const float*)d_in[0];
    const float* w_v   = (const float*)d_in[1];
    const float* b_v   = (const float*)d_in[2];
    const float* g_vn  = (const float*)d_in[3];
    const float* b_vn  = (const float*)d_in[4];
    const float* w_k   = (const float*)d_in[5];
    const float* b_k   = (const float*)d_in[6];
    const float* g_kn  = (const float*)d_in[7];
    const float* b_kn  = (const float*)d_in[8];
    const float* w_fc  = (const float*)d_in[9];
    const float* b_fc  = (const float*)d_in[10];
    const float* g_fcn = (const float*)d_in[11];
    const float* b_fcn = (const float*)d_in[12];
    float* out = (float*)d_out;

    char* ws = (char*)d_ws;
    size_t off = 0;
    auto alloc = [&](size_t bytes) {
        void* p = ws + off;
        off = (off + bytes + 255) & ~(size_t)255;
        return p;
    };
    __bf16* vkpre  = (__bf16*)alloc((size_t)MROWS * NCOLS * sizeof(__bf16));
    float*  sums1  = (float*) alloc(NCOLS * sizeof(float));
    float*  sumsq1 = (float*) alloc(NCOLS * sizeof(float));
    float*  attn   = (float*) alloc((size_t)BATCH * HEADS * NPIX * sizeof(float));
    float*  ma     = (float*) alloc((size_t)BATCH * HEADS * NWIN * sizeof(float));
    float*  ybuf   = (float*) alloc((size_t)BATCH * NROWS2 * CCH * sizeof(float));
    __bf16* x0bf   = (__bf16*)alloc((size_t)BATCH * CIN * sizeof(__bf16));
    __bf16* yy     = (__bf16*)alloc((size_t)MFC * KFC * sizeof(__bf16));
    float*  fc     = (float*) alloc((size_t)MFC * CCH * sizeof(float));
    float*  sums2  = (float*) alloc(CCH * sizeof(float));
    float*  sumsq2 = (float*) alloc(CCH * sizeof(float));

    gemm_vk_kernel<<<dim3(MROWS / 64, NCOLS / 128), 256, 0, stream>>>(
        x, w_v, w_k, b_v, b_k, vkpre);
    colstats_bf16_kernel<<<2, 256, 0, stream>>>(
        vkpre, MROWS, NCOLS / 2, sums1, sumsq1);
    attn_softmax_kernel<<<BATCH * HEADS, 256, 0, stream>>>(
        vkpre, sums1, sumsq1, g_kn, b_kn, attn);
    ma_calc_kernel<<<BATCH * HEADS, 256, 0, stream>>>(attn, ma);
    xw_y_kernel<<<dim3(CCH / 8, BATCH), 256, 0, stream>>>(
        vkpre, sums1, sumsq1, g_vn, b_vn, attn, ma, ybuf);
    x0_extract_kernel<<<(BATCH * CIN + 255) / 256, 256, 0, stream>>>(x, x0bf);
    ynorm_yy_kernel<<<MFC, 256, 0, stream>>>(x0bf, ybuf, yy);
    gemm_fc_kernel<<<dim3(MFC / 64, CCH / 128), 256, 0, stream>>>(
        yy, w_fc, b_fc, fc);
    colstats_f32_kernel<<<CCH / 256, 256, 0, stream>>>(
        fc, MFC, CCH, sums2, sumsq2);
    bn_relu_out_kernel<<<(MFC * CCH + 255) / 256, 256, 0, stream>>>(
        fc, sums2, sumsq2, g_fcn, b_fcn, out);
}
